// GNN_graphCL_46059229282626
// MI455X (gfx1250) — matrix operand order repaired and checked by reading, not compiled, not run on hardware
//
#include <hip/hip_runtime.h>
#include <hip/hip_bf16.h>

// ---------------------------------------------------------------------------
// GIN / GraphCL forward for MI455X (gfx1250, wave32, WMMA bf16)
// ---------------------------------------------------------------------------

#define N_NODES  50000
#define N_EDGES  200000
#define N_GRAPHS 2048
#define EMB      512
#define N_LAYERS 5

typedef __bf16 bf16_t;
typedef bf16_t  bf16x16 __attribute__((ext_vector_type(16)));
typedef float   f32x8   __attribute__((ext_vector_type(8)));
typedef unsigned int u32x8 __attribute__((ext_vector_type(8)));

union FragCast { u32x8 u; bf16x16 b; };

__device__ __forceinline__ unsigned short f32_to_bf16_rne(float f) {
    unsigned int u = __float_as_uint(f);
    u += 0x7FFFu + ((u >> 16) & 1u);          // round-to-nearest-even
    return (unsigned short)(u >> 16);
}

// ---------------------------------------------------------------------------
// f32 -> bf16 bulk convert (weights)
// ---------------------------------------------------------------------------
__global__ void cvt_bf16_kernel(const float* __restrict__ src,
                                unsigned short* __restrict__ dst, int n) {
    int i = blockIdx.x * blockDim.x + threadIdx.x;
    if (i < n) dst[i] = f32_to_bf16_rne(src[i]);
}

// ---------------------------------------------------------------------------
// AtomEncoder: h[i] = atom_emb1[x[i,0]] + atom_emb2[x[i,1]]
// ---------------------------------------------------------------------------
__global__ __launch_bounds__(128)
void atom_encode_kernel(const int* __restrict__ x,
                        const float* __restrict__ e1,
                        const float* __restrict__ e2,
                        float* __restrict__ h) {
    int i = blockIdx.x;
    int d = threadIdx.x * 4;
    int a = x[i * 2 + 0];
    int c = x[i * 2 + 1];
    float4 v1 = *(const float4*)&e1[(size_t)a * EMB + d];
    float4 v2 = *(const float4*)&e2[(size_t)c * EMB + d];
    float4 o; o.x = v1.x + v2.x; o.y = v1.y + v2.y; o.z = v1.z + v2.z; o.w = v1.w + v2.w;
    *(float4*)&h[(size_t)i * EMB + d] = o;
}

// ---------------------------------------------------------------------------
// GIN message + scatter-add:  agg[dst] += h[src] + bond1[ea0] + bond2[ea1]
// blocks 0..E-1 = real edges, blocks E..E+N-1 = self loops (attr 6,3)
// ---------------------------------------------------------------------------
__global__ __launch_bounds__(128)
void scatter_kernel(const float* __restrict__ h,
                    const int* __restrict__ ei,     // [2,E]
                    const int* __restrict__ ea,     // [E,2]
                    const float* __restrict__ bond1,
                    const float* __restrict__ bond2,
                    float* __restrict__ agg, int E) {
    int e = blockIdx.x;
    int d = threadIdx.x * 4;
    int src, dst, a0, a1;
    if (e < E) {
        src = ei[e]; dst = ei[E + e];
        a0 = ea[e * 2 + 0]; a1 = ea[e * 2 + 1];
    } else {
        int j = e - E; src = j; dst = j; a0 = 6; a1 = 3;
    }
    float4 hv = *(const float4*)&h[(size_t)src * EMB + d];
    float4 b1 = *(const float4*)&bond1[(size_t)a0 * EMB + d];
    float4 b2 = *(const float4*)&bond2[(size_t)a1 * EMB + d];
    float* out = &agg[(size_t)dst * EMB + d];
    atomicAdd(out + 0, hv.x + b1.x + b2.x);
    atomicAdd(out + 1, hv.y + b1.y + b2.y);
    atomicAdd(out + 2, hv.z + b1.z + b2.z);
    atomicAdd(out + 3, hv.w + b1.w + b2.w);
}

// ---------------------------------------------------------------------------
// WMMA bf16 GEMM:  C[M,N] = act(A[M,K] @ B[K,N] + bias[N])
//   AT = float (convert to bf16 during staging) or unsigned short (bf16 bits)
//   OUT_BF16: store C as bf16 bits; else fp32
// Block tile 128x64x32, 256 threads = 8 waves; each wave owns a 16x64 stripe
// -> A fragment reused across 4 v_wmma_f32_16x16x32_bf16 per k-step.
// Software-pipelined: global->regs for tile k+1 overlaps WMMA of tile k;
// LDS double-buffered; branch-free OOB via row clamp + select.
// All 4 B fragments are loaded before the WMMA sequence so LDS latency
// overlaps the matrix pipe instead of serializing on s_wait_dscnt 0.
// ---------------------------------------------------------------------------
template <typename AT, bool OUT_BF16, bool RELU>
__global__ __launch_bounds__(256)
void gemm_kernel(const AT* __restrict__ A,
                 const unsigned short* __restrict__ B,
                 const float* __restrict__ bias,
                 void* __restrict__ Cout,
                 int M, int N, int K) {
    constexpr int BM = 128, BN = 64, BK = 32;
    constexpr int LDA = BK + 2;          // even -> 32-bit LDS accesses aligned
    constexpr int LDB = BK + 2;
    __shared__ unsigned short As[2][BM * LDA];   // [m][k]
    __shared__ unsigned short Bs[2][BN * LDB];   // [n][k] (transposed stage)

    const int tid  = threadIdx.x;
    const int lane = tid & 31;
    const int wave = tid >> 5;           // 0..7  = M subtile of this wave
    const int bm   = blockIdx.y * BM;
    const int bn   = blockIdx.x * BN;
    const int l15  = lane & 15;
    const int hh   = lane >> 4;

    // staging coordinates (fixed per thread)
    const int a_r0 = tid >> 3;           // A: rows a_r0 + {0,32,64,96}
    const int a_c4 = (tid & 7) * 4;      // A: 4 consecutive k
    const int b_k  = tid >> 3;           // B: one k-row
    const int b_n0 = (tid & 7) * 8;      // B: 8 consecutive n

    float4  aF[4];
    ushort4 aH[4];
    bool    aOk[4];
    uint4   bR;

    auto load_tile = [&](int kb) {
        #pragma unroll
        for (int i = 0; i < 4; ++i) {
            int gr  = bm + a_r0 + i * 32;
            aOk[i]  = gr < M;
            int grc = aOk[i] ? gr : (M - 1);     // clamp: always-legal load
            if constexpr (sizeof(AT) == 4)
                aF[i] = *(const float4*)&((const float*)A)[(size_t)grc * K + kb + a_c4];
            else
                aH[i] = *(const ushort4*)&((const unsigned short*)A)[(size_t)grc * K + kb + a_c4];
        }
        bR = *(const uint4*)&B[(size_t)(kb + b_k) * N + bn + b_n0];
    };

    auto store_tile = [&](int buf) {
        #pragma unroll
        for (int i = 0; i < 4; ++i) {
            unsigned short s0, s1, s2, s3;
            if constexpr (sizeof(AT) == 4) {
                s0 = f32_to_bf16_rne(aF[i].x); s1 = f32_to_bf16_rne(aF[i].y);
                s2 = f32_to_bf16_rne(aF[i].z); s3 = f32_to_bf16_rne(aF[i].w);
            } else {
                s0 = aH[i].x; s1 = aH[i].y; s2 = aH[i].z; s3 = aH[i].w;
            }
            unsigned int lo = aOk[i] ? ((unsigned int)s0 | ((unsigned int)s1 << 16)) : 0u;
            unsigned int hi = aOk[i] ? ((unsigned int)s2 | ((unsigned int)s3 << 16)) : 0u;
            unsigned int* p = (unsigned int*)&As[buf][(a_r0 + i * 32) * LDA + a_c4];
            p[0] = lo; p[1] = hi;
        }
        unsigned int w[4] = {bR.x, bR.y, bR.z, bR.w};
        #pragma unroll
        for (int j = 0; j < 4; ++j) {
            Bs[buf][(b_n0 + 2 * j    ) * LDB + b_k] = (unsigned short)(w[j] & 0xFFFFu);
            Bs[buf][(b_n0 + 2 * j + 1) * LDB + b_k] = (unsigned short)(w[j] >> 16);
        }
    };

    f32x8 acc[4] = {};

    auto compute = [&](int buf) {
        FragCast fa;
        #pragma unroll
        for (int v = 0; v < 8; ++v) {     // ISA §7.12.2: 16-bit A 16x32 layout
            int k0 = ((v & 3) * 2) + ((v >> 2) * 16) + hh * 8;
            fa.u[v] = *(const unsigned int*)&As[buf][(wave * 16 + l15) * LDA + k0];
        }
        FragCast fb[4];
        #pragma unroll
        for (int t = 0; t < 4; ++t) {     // issue ALL fragment loads first
            #pragma unroll
            for (int v = 0; v < 8; ++v) { // B 32x16: same k order as A
                int k0 = ((v & 3) * 2) + ((v >> 2) * 16) + hh * 8;
                fb[t].u[v] = *(const unsigned int*)&Bs[buf][(t * 16 + l15) * LDB + k0];
            }
        }
        #pragma unroll
        for (int t = 0; t < 4; ++t)       // then back-to-back WMMAs
            acc[t] = __builtin_amdgcn_wmma_f32_16x16x32_bf16(
                false, fa.b, false, fb[t].b, (short)0, acc[t], false, false);
    };

    // ---- software-pipelined main loop ----
    const int nK = K / BK;
    load_tile(0);
    store_tile(0);
    __syncthreads();
    for (int kt = 0; kt < nK; ++kt) {
        const int cur = kt & 1;
        if (kt + 1 < nK) load_tile((kt + 1) * BK);   // global loads in flight...
        compute(cur);                                 // ...while WMMA runs
        if (kt + 1 < nK) store_tile(cur ^ 1);
        __syncthreads();
    }

    // ---- epilogue: bias (+ReLU), store ----
    float bv[4];
    #pragma unroll
    for (int t = 0; t < 4; ++t) bv[t] = bias[bn + t * 16 + l15];
    #pragma unroll
    for (int v = 0; v < 8; ++v) {
        int row = bm + wave * 16 + v + 8 * hh;
        if (row < M) {
            #pragma unroll
            for (int t = 0; t < 4; ++t) {
                float xv = acc[t][v] + bv[t];
                if (RELU) xv = fmaxf(xv, 0.0f);
                int col = bn + t * 16 + l15;
                if constexpr (OUT_BF16)
                    ((unsigned short*)Cout)[(size_t)row * N + col] = f32_to_bf16_rne(xv);
                else
                    ((float*)Cout)[(size_t)row * N + col] = xv;
            }
        }
    }
}

// ---------------------------------------------------------------------------
// Mean pooling over graphs
// ---------------------------------------------------------------------------
__global__ __launch_bounds__(128)
void pool_scatter_kernel(const float* __restrict__ h,
                         const int* __restrict__ batch,
                         float* __restrict__ pooled,
                         float* __restrict__ cnt) {
    int i = blockIdx.x;
    int d = threadIdx.x * 4;
    int g = batch[i];
    float4 v = *(const float4*)&h[(size_t)i * EMB + d];
    float* out = &pooled[(size_t)g * EMB + d];
    atomicAdd(out + 0, v.x);
    atomicAdd(out + 1, v.y);
    atomicAdd(out + 2, v.z);
    atomicAdd(out + 3, v.w);
    if (threadIdx.x == 0) atomicAdd(&cnt[g], 1.0f);
}

__global__ __launch_bounds__(128)
void pool_div_kernel(float* __restrict__ pooled, const float* __restrict__ cnt) {
    int g = blockIdx.x;
    int d = threadIdx.x * 4;
    float inv = 1.0f / fmaxf(cnt[g], 1.0f);
    float4 v = *(float4*)&pooled[(size_t)g * EMB + d];
    v.x *= inv; v.y *= inv; v.z *= inv; v.w *= inv;
    *(float4*)&pooled[(size_t)g * EMB + d] = v;
}

// ---------------------------------------------------------------------------
// Orchestration
// ---------------------------------------------------------------------------
extern "C" void kernel_launch(void* const* d_in, const int* in_sizes, int n_in,
                              void* d_out, int out_size, void* d_ws, size_t ws_size,
                              hipStream_t stream) {
    const int*   x     = (const int*)  d_in[0];
    const int*   ei    = (const int*)  d_in[1];
    const int*   ea    = (const int*)  d_in[2];
    const int*   batch = (const int*)  d_in[3];
    const float* ae1   = (const float*)d_in[4];
    const float* ae2   = (const float*)d_in[5];
    const float* be1   = (const float*)d_in[6];   // [L,8,512]
    const float* be2   = (const float*)d_in[7];   // [L,4,512]
    const float* W1    = (const float*)d_in[8];   // [L,512,1024]
    const float* b1    = (const float*)d_in[9];   // [L,1024]
    const float* W2    = (const float*)d_in[10];  // [L,1024,512]
    const float* b2    = (const float*)d_in[11];  // [L,512]
    const float* pW1   = (const float*)d_in[12];  // [512,512]
    const float* pb1   = (const float*)d_in[13];
    const float* pW2   = (const float*)d_in[14];
    const float* pb2   = (const float*)d_in[15];
    float* out = (float*)d_out;                   // [2048,512]

    // ---- workspace carve-up (256B aligned) ----
    size_t off = 0;
    auto alloc = [&](size_t bytes) -> void* {
        void* p = (char*)d_ws + off;
        off += (bytes + 255) & ~(size_t)255;
        return p;
    };
    float*          h      = (float*)         alloc((size_t)N_NODES * EMB * 4);
    float*          agg    = (float*)         alloc((size_t)N_NODES * EMB * 4);
    unsigned short* t      = (unsigned short*)alloc((size_t)N_NODES * 2 * EMB * 2);
    unsigned short* Wb1    = (unsigned short*)alloc((size_t)N_LAYERS * EMB * 2 * EMB * 2);
    unsigned short* Wb2    = (unsigned short*)alloc((size_t)N_LAYERS * 2 * EMB * EMB * 2);
    unsigned short* pWb1   = (unsigned short*)alloc((size_t)EMB * EMB * 2);
    unsigned short* pWb2   = (unsigned short*)alloc((size_t)EMB * EMB * 2);
    float*          pooled = (float*)         alloc((size_t)N_GRAPHS * EMB * 4);
    float*          cnt    = (float*)         alloc((size_t)N_GRAPHS * 4);
    unsigned short* q      = (unsigned short*)alloc((size_t)N_GRAPHS * EMB * 2);

    // ---- convert weights to bf16 (deterministic every call) ----
    {
        int n1 = N_LAYERS * EMB * 2 * EMB;
        cvt_bf16_kernel<<<(n1 + 255) / 256, 256, 0, stream>>>(W1, Wb1, n1);
        cvt_bf16_kernel<<<(n1 + 255) / 256, 256, 0, stream>>>(W2, Wb2, n1);
        int n2 = EMB * EMB;
        cvt_bf16_kernel<<<(n2 + 255) / 256, 256, 0, stream>>>(pW1, pWb1, n2);
        cvt_bf16_kernel<<<(n2 + 255) / 256, 256, 0, stream>>>(pW2, pWb2, n2);
    }

    // ---- atom encoder ----
    atom_encode_kernel<<<N_NODES, 128, 0, stream>>>(x, ae1, ae2, h);

    const dim3 blk(256);
    const dim3 g1((2 * EMB) / 64, (N_NODES + 127) / 128);  // 50000 x 1024
    const dim3 g2(EMB / 64,       (N_NODES + 127) / 128);  // 50000 x 512

    for (int l = 0; l < N_LAYERS; ++l) {
        hipMemsetAsync(agg, 0, (size_t)N_NODES * EMB * 4, stream);
        scatter_kernel<<<N_EDGES + N_NODES, 128, 0, stream>>>(
            h, ei, ea,
            be1 + (size_t)l * 8 * EMB,
            be2 + (size_t)l * 4 * EMB,
            agg, N_EDGES);
        // t = relu(agg @ W1 + b1)   (bf16 out)
        gemm_kernel<float, true, true><<<g1, blk, 0, stream>>>(
            agg, Wb1 + (size_t)l * EMB * 2 * EMB, b1 + (size_t)l * 2 * EMB,
            t, N_NODES, 2 * EMB, EMB);
        // h = [relu](t @ W2 + b2)   (f32 out; relu only for l < L-1)
        if (l < N_LAYERS - 1)
            gemm_kernel<unsigned short, false, true><<<g2, blk, 0, stream>>>(
                t, Wb2 + (size_t)l * 2 * EMB * EMB, b2 + (size_t)l * EMB,
                h, N_NODES, EMB, 2 * EMB);
        else
            gemm_kernel<unsigned short, false, false><<<g2, blk, 0, stream>>>(
                t, Wb2 + (size_t)l * 2 * EMB * EMB, b2 + (size_t)l * EMB,
                h, N_NODES, EMB, 2 * EMB);
    }

    // ---- mean pooling ----
    hipMemsetAsync(pooled, 0, (size_t)N_GRAPHS * EMB * 4, stream);
    hipMemsetAsync(cnt,    0, (size_t)N_GRAPHS * 4,       stream);
    pool_scatter_kernel<<<N_NODES, 128, 0, stream>>>(h, batch, pooled, cnt);
    pool_div_kernel<<<N_GRAPHS, 128, 0, stream>>>(pooled, cnt);

    // ---- projection head ----
    const dim3 gp(EMB / 64, N_GRAPHS / 128);
    gemm_kernel<float, true, true><<<gp, blk, 0, stream>>>(
        pooled, pWb1, pb1, q, N_GRAPHS, EMB, EMB);
    gemm_kernel<unsigned short, false, false><<<gp, blk, 0, stream>>>(
        q, pWb2, pb2, out, N_GRAPHS, EMB, EMB);
}
